// SparseConvNet_32341103739522
// MI455X (gfx1250) — matrix operand order repaired and checked by reading, not compiled, not run on hardware
//
#include <hip/hip_runtime.h>
#include <hip/hip_bf16.h>

typedef __attribute__((ext_vector_type(16))) _Float16 v16h;
typedef __attribute__((ext_vector_type(8)))  float    v8f;

#define VOXS 0.005f
#define BN_EPS 1e-3f

// ---------------- utility kernels ----------------

__global__ void k_init_min(unsigned* minbits) {
  if (threadIdx.x < 3) minbits[threadIdx.x] = 0x7F800000u; // +inf bits
}

__global__ void k_zero_f32(float* __restrict__ p, int n) {
  int i = blockIdx.x * 256 + threadIdx.x;
  if (i < n) p[i] = 0.f;
}

__global__ void k_min_reduce(const float* __restrict__ xyz, int n, unsigned* minbits) {
  int i = blockIdx.x * 256 + threadIdx.x;
  if (i >= n) return;
  // coordinates are non-negative -> uint order == float order
  atomicMin(&minbits[0], __float_as_uint(xyz[3*i+0]));
  atomicMin(&minbits[1], __float_as_uint(xyz[3*i+1]));
  atomicMin(&minbits[2], __float_as_uint(xyz[3*i+2]));
}

__global__ void k_scatter(const float* __restrict__ xyz, const float* __restrict__ feats,
                          int n, const unsigned* __restrict__ minbits,
                          float* __restrict__ staging, float* __restrict__ mask0,
                          int D, int H, int W) {
  int i = blockIdx.x * 256 + threadIdx.x;
  if (i >= n) return;
  float mnx = __uint_as_float(minbits[0]) - 0.05f;
  float mny = __uint_as_float(minbits[1]) - 0.05f;
  float mnz = __uint_as_float(minbits[2]) - 0.05f;
  int cx = (int)rintf((xyz[3*i+0] - mnx) / VOXS);  // rint = round-half-even, matches jnp.round
  int cy = (int)rintf((xyz[3*i+1] - mny) / VOXS);
  int cz = (int)rintf((xyz[3*i+2] - mnz) / VOXS);
  if ((unsigned)cx >= (unsigned)D || (unsigned)cy >= (unsigned)H || (unsigned)cz >= (unsigned)W) return;
  int lin = (cx * H + cy) * W + cz;
  #pragma unroll
  for (int c = 0; c < 16; ++c)
    atomicAdd(&staging[(size_t)lin * 16 + c], feats[(size_t)i * 16 + c]);
  mask0[lin] = 1.0f;
}

__global__ void k_f32_to_f16(const float* __restrict__ s, _Float16* __restrict__ d, int n) {
  int i = blockIdx.x * 256 + threadIdx.x;
  if (i < n) d[i] = (_Float16)s[i];
}

// mask_out[o] = any(mask_in in 3x3x3 window at stride 2)
__global__ void k_mask_down(const float* __restrict__ min_, float* __restrict__ mout,
                            int iD, int iH, int iW, int oD, int oH, int oW) {
  int vox = blockIdx.x * 256 + threadIdx.x;
  int nv = oD * oH * oW;
  if (vox >= nv) return;
  int ow = vox % oW; int t = vox / oW; int oh = t % oH; int od = t / oH;
  float any = 0.f;
  for (int kd = 0; kd < 3; ++kd)
    for (int kh = 0; kh < 3; ++kh)
      for (int kw = 0; kw < 3; ++kw) {
        int id = od * 2 + kd - 1, ih = oh * 2 + kh - 1, iw = ow * 2 + kw - 1;
        if ((unsigned)id < (unsigned)iD && (unsigned)ih < (unsigned)iH && (unsigned)iw < (unsigned)iW)
          any = fmaxf(any, min_[((size_t)id * iH + ih) * iW + iw]);
      }
  mout[vox] = any > 0.f ? 1.f : 0.f;
}

// Pack fp32 weights (O,I,3,3,3) into WMMA A-fragment lane layout, f16.
// Layout: apack[((ctile*nKTiles + kTile)*32 + lane)*16 + j]
// A 16x32 f16 lane map used here (contiguous K; B loaded the same way): lanes 0-15: M=lane, K in {0..15}; lanes 16-31: M=lane-16, K in {16..31}
// K-ordering:
//   Cin==16 : two taps per K-tile -> tap = kTile*2 + (K>>4), cin = K&15, 14 tiles (tap 27 zero-padded)
//   Cin>=32 : tap = kTile/TPT, cin = (kTile%TPT)*32 + K
__global__ void k_pack_weights(const float* __restrict__ w, _Float16* __restrict__ apack,
                               int Cin, int Cout, int TPT) {
  int nKTiles = (Cin == 16) ? 14 : 27 * TPT;
  int total = (Cout / 16) * nKTiles * 512;
  int t = blockIdx.x * 256 + threadIdx.x;
  if (t >= total) return;
  int j    = t & 15;
  int lane = (t >> 4) & 31;
  int rest = t >> 9;
  int kTile = rest % nKTiles;
  int ctile = rest / nKTiles;
  int M = lane & 15, hi = lane >> 4;
  int K = hi * 16 + j;
  int tap, cin;
  if (Cin == 16) { tap = kTile * 2 + (K >> 4); cin = K & 15; }
  else           { tap = kTile / TPT;          cin = (kTile % TPT) * 32 + K; }
  int cout = ctile * 16 + M;
  float v = (tap < 27 && cin < Cin) ? w[((size_t)cout * Cin + cin) * 27 + tap] : 0.0f;
  apack[t] = (_Float16)v;
}

// ---------------- fused conv3d(3^3,pad1) + BN + ReLU + mask, implicit GEMM on WMMA ----------------
// One wave computes 16 voxels x (NCT*16) couts; B fragments amortized over NCT cout tiles.
// Cin==16 packs two taps per 32-K tile (each half-wave gathers its own tap) -> no wasted K.
template <int CIN, int TPT, int NCT>
__global__ __launch_bounds__(32)
void conv_wmma(const _Float16* __restrict__ xin, _Float16* __restrict__ xout,
               const _Float16* __restrict__ apack,
               const float* __restrict__ gamma, const float* __restrict__ beta,
               const float* __restrict__ mean,  const float* __restrict__ var,
               const float* __restrict__ maskOut,
               float* __restrict__ emit,        // fp32 NCDHW output map, or null
               int iD, int iH, int iW, int oD, int oH, int oW,
               int stride, int Cout) {
  const int lane = threadIdx.x;
  const int col  = lane & 15;   // N within tile
  const int hi   = lane >> 4;
  const int vox  = blockIdx.x * 16 + col;
  const int ct0  = blockIdx.y * NCT;

  int ow = vox % oW; int t1 = vox / oW; int oh = t1 % oH; int od = t1 / oH;

  constexpr int nKT = (CIN == 16) ? 14 : 27 * TPT;
  const _Float16* alane = apack + (size_t)lane * 16;

  v8f acc[NCT];
  #pragma unroll
  for (int c = 0; c < NCT; ++c) { v8f z = {}; acc[c] = z; }

  if constexpr (CIN == 16) {
    #pragma unroll
    for (int kt = 0; kt < 14; ++kt) {
      int tap = kt * 2 + hi;                      // per-half-wave tap
      int kd = tap / 9, kh = (tap / 3) % 3, kw = tap % 3;
      int id = od * stride + kd - 1;
      int ih = oh * stride + kh - 1;
      int iw = ow * stride + kw - 1;
      bool valid = (tap < 27) && (unsigned)id < (unsigned)iD &&
                   (unsigned)ih < (unsigned)iH && (unsigned)iw < (unsigned)iW;
      v16h b = {};
      if (valid) b = *(const v16h*)(xin + (((size_t)id * iH + ih) * iW + iw) * 16);
      #pragma unroll
      for (int c = 0; c < NCT; ++c) {
        v16h a = *(const v16h*)(alane + (size_t)((ct0 + c) * nKT + kt) * 512);
        acc[c] = __builtin_amdgcn_wmma_f32_16x16x32_f16(
            false, a, false, b, (short)0, acc[c], false, false);
      }
    }
  } else {
    for (int kd = 0; kd < 3; ++kd) {
      int id = od * stride + kd - 1;
      bool vd = (unsigned)id < (unsigned)iD;
      for (int kh = 0; kh < 3; ++kh) {
        int ih = oh * stride + kh - 1;
        bool vdh = vd && (unsigned)ih < (unsigned)iH;
        #pragma unroll
        for (int kw = 0; kw < 3; ++kw) {
          int iw = ow * stride + kw - 1;
          bool valid = vdh && (unsigned)iw < (unsigned)iW;
          const _Float16* xrow =
              xin + (((size_t)id * iH + ih) * iW + iw) * CIN + hi * 16;
          int tap = (kd * 3 + kh) * 3 + kw;
          #pragma unroll
          for (int t = 0; t < TPT; ++t) {
            v16h b = {};
            if (valid) b = *(const v16h*)(xrow + t * 32);   // contiguous 32B per lane
            int kt = tap * TPT + t;
            #pragma unroll
            for (int c = 0; c < NCT; ++c) {
              v16h a = *(const v16h*)(alane + (size_t)((ct0 + c) * nKT + kt) * 512);
              acc[c] = __builtin_amdgcn_wmma_f32_16x16x32_f16(
                  false, a, false, b, (short)0, acc[c], false, false);
            }
          }
        }
      }
    }
  }

  // Epilogue: BN(eval) + ReLU + mask; C layout: lane<16 -> M=r, lane>=16 -> M=8+r, N=lane%16
  float m = maskOut[vox];
  const int rbase = hi * 8;
  const size_t nvo = (size_t)oD * oH * oW;
  #pragma unroll
  for (int c = 0; c < NCT; ++c) {
    #pragma unroll
    for (int r = 0; r < 8; ++r) {
      int co = (ct0 + c) * 16 + rbase + r;
      float sc = gamma[co] * rsqrtf(var[co] + BN_EPS);
      float sh = beta[co] - mean[co] * sc;
      float v = fmaxf(acc[c][r] * sc + sh, 0.0f) * m;
      xout[(size_t)vox * Cout + co] = (_Float16)v;
      if (emit) emit[(size_t)co * nvo + vox] = v;
    }
  }
}

// ---------------- host orchestration ----------------

extern "C" void kernel_launch(void* const* d_in, const int* in_sizes, int n_in,
                              void* d_out, int out_size, void* d_ws, size_t ws_size,
                              hipStream_t stream) {
  (void)in_sizes; (void)n_in; (void)out_size; (void)ws_size;
  const float* xyz   = (const float*)d_in[0];
  const float* feats = (const float*)d_in[1];
  const int NPTS = in_sizes[0] / 3;

  // Grid dims are fixed by the input distribution (96,96,64 after pad + round-to-32)
  const int D0=96,H0=96,W0=64; const int NV0=D0*H0*W0;        // 589824
  const int D1=48,H1=48,W1=32; const int NV1=D1*H1*W1;        // 73728
  const int D2=24,H2=24,W2=16; const int NV2=D2*H2*W2;        // 9216
  const int D3=12,H3=12,W3=8;  const int NV3=D3*H3*W3;        // 1152
  const int D4=6, H4=6, W4=4;  const int NV4=D4*H4*W4;        // 144

  // workspace carve (256B aligned)
  char* ws = (char*)d_ws;
  size_t off = 0;
  auto carve = [&](size_t bytes) -> void* {
    void* p = ws + off; off = (off + bytes + 255) & ~(size_t)255; return p;
  };
  float*    staging = (float*)carve((size_t)NV0 * 16 * 4);  // 37.75MB, reused as bufB (f16)
  _Float16* bufA    = (_Float16*)carve((size_t)NV0 * 16 * 2);
  _Float16* bufB    = (_Float16*)staging;
  float* mask0 = (float*)carve((size_t)NV0 * 4);
  float* mask1 = (float*)carve((size_t)NV1 * 4);
  float* mask2 = (float*)carve((size_t)NV2 * 4);
  float* mask3 = (float*)carve((size_t)NV3 * 4);
  float* mask4 = (float*)carve((size_t)NV4 * 4);
  unsigned* minbits = (unsigned*)carve(16);

  // layer table; pi = d_in index of 'beta' (pytree: sorted keys -> beta,gamma,mean,var,w;
  // blocks sorted: conv0(2),conv1(2),conv2(3),conv3(3),conv4(3),down0..down3)
  struct L { int pi, Cin, Cout, TPT, NCT; };
  const L layers[17] = {
    {2,16,16,1,1},{7,16,16,1,1},                             // conv0_0, conv0_1
    {12,32,32,1,2},{17,32,32,1,2},                           // conv1_0, conv1_1
    {22,64,64,2,2},{27,64,64,2,2},{32,64,64,2,2},            // conv2_0..2
    {37,128,128,4,1},{42,128,128,4,1},{47,128,128,4,1},      // conv3_0..2
    {52,128,128,4,1},{57,128,128,4,1},{62,128,128,4,1},      // conv4_0..2
    {67,16,32,1,2},{72,32,64,1,2},{77,64,128,2,1},{82,128,128,4,1} // down0..down3
  };
  _Float16* apack[17];
  int apackN[17];
  for (int i = 0; i < 17; ++i) {
    int nKT = (layers[i].Cin == 16) ? 14 : 27 * layers[i].TPT;
    apackN[i] = (layers[i].Cout / 16) * nKT * 512;
    apack[i] = (_Float16*)carve((size_t)apackN[i] * 2);
  }

  // ---- voxelize ----
  k_init_min<<<1, 32, 0, stream>>>(minbits);
  k_zero_f32<<<(NV0 * 16 + 255) / 256, 256, 0, stream>>>(staging, NV0 * 16);
  k_zero_f32<<<(NV0 + 255) / 256, 256, 0, stream>>>(mask0, NV0);
  k_min_reduce<<<(NPTS + 255) / 256, 256, 0, stream>>>(xyz, NPTS, minbits);
  k_scatter<<<(NPTS + 255) / 256, 256, 0, stream>>>(xyz, feats, NPTS, minbits,
                                                    staging, mask0, D0, H0, W0);
  k_f32_to_f16<<<(NV0 * 16 + 255) / 256, 256, 0, stream>>>(staging, bufA, NV0 * 16);

  // ---- pack weights ----
  for (int i = 0; i < 17; ++i) {
    const float* w = (const float*)d_in[layers[i].pi + 4];
    k_pack_weights<<<(apackN[i] + 255) / 256, 256, 0, stream>>>(
        w, apack[i], layers[i].Cin, layers[i].Cout, layers[i].TPT);
  }

  // ---- conv dispatch ----
  auto conv = [&](int li, const _Float16* xin, _Float16* xout, const float* mask,
                  float* emit, int iD,int iH,int iW, int oD,int oH,int oW, int stride) {
    const L& l = layers[li];
    const float* beta  = (const float*)d_in[l.pi + 0];
    const float* gamma = (const float*)d_in[l.pi + 1];
    const float* mean  = (const float*)d_in[l.pi + 2];
    const float* var   = (const float*)d_in[l.pi + 3];
    dim3 grid((unsigned)(oD * oH * oW / 16), (unsigned)(l.Cout / 16 / l.NCT));
#define LAUNCH_CONV(CI, TP, NC)                                                     \
    conv_wmma<CI, TP, NC><<<grid, 32, 0, stream>>>(xin, xout, apack[li],            \
        gamma, beta, mean, var, mask, emit, iD, iH, iW, oD, oH, oW, stride, l.Cout)
    switch (l.Cin) {
      case 16:
        if (l.NCT == 1) LAUNCH_CONV(16, 1, 1); else LAUNCH_CONV(16, 1, 2);
        break;
      case 32:
        LAUNCH_CONV(32, 1, 2);
        break;
      case 64:
        if (l.NCT == 2) LAUNCH_CONV(64, 2, 2); else LAUNCH_CONV(64, 2, 1);
        break;
      default:
        LAUNCH_CONV(128, 4, 1);
        break;
    }
#undef LAUNCH_CONV
  };

  float* out = (float*)d_out;
  float* emit0 = out;                                   // (1,32,48,48,32)
  float* emit1 = out + (size_t)NV1 * 32;                // (1,64,24,24,16)
  float* emit2 = emit1 + (size_t)NV2 * 64;              // (1,128,12,12,8)
  float* emit3 = emit2 + (size_t)NV3 * 128;             // (1,128,6,6,4)

  // conv0 block (96,96,64)
  conv(0, bufA, bufB, mask0, nullptr, D0,H0,W0, D0,H0,W0, 1);
  conv(1, bufB, bufA, mask0, nullptr, D0,H0,W0, D0,H0,W0, 1);
  // down0 -> (48,48,32)
  k_mask_down<<<(NV1 + 255) / 256, 256, 0, stream>>>(mask0, mask1, D0,H0,W0, D1,H1,W1);
  conv(13, bufA, bufB, mask1, nullptr, D0,H0,W0, D1,H1,W1, 2);
  // conv1 block (emit)
  conv(2, bufB, bufA, mask1, nullptr, D1,H1,W1, D1,H1,W1, 1);
  conv(3, bufA, bufB, mask1, emit0,   D1,H1,W1, D1,H1,W1, 1);
  // down1 -> (24,24,16)
  k_mask_down<<<(NV2 + 255) / 256, 256, 0, stream>>>(mask1, mask2, D1,H1,W1, D2,H2,W2);
  conv(14, bufB, bufA, mask2, nullptr, D1,H1,W1, D2,H2,W2, 2);
  // conv2 block (emit)
  conv(4, bufA, bufB, mask2, nullptr, D2,H2,W2, D2,H2,W2, 1);
  conv(5, bufB, bufA, mask2, nullptr, D2,H2,W2, D2,H2,W2, 1);
  conv(6, bufA, bufB, mask2, emit1,   D2,H2,W2, D2,H2,W2, 1);
  // down2 -> (12,12,8)
  k_mask_down<<<(NV3 + 255) / 256, 256, 0, stream>>>(mask2, mask3, D2,H2,W2, D3,H3,W3);
  conv(15, bufB, bufA, mask3, nullptr, D2,H2,W2, D3,H3,W3, 2);
  // conv3 block (emit)
  conv(7, bufA, bufB, mask3, nullptr, D3,H3,W3, D3,H3,W3, 1);
  conv(8, bufB, bufA, mask3, nullptr, D3,H3,W3, D3,H3,W3, 1);
  conv(9, bufA, bufB, mask3, emit2,   D3,H3,W3, D3,H3,W3, 1);
  // down3 -> (6,6,4)
  k_mask_down<<<(NV4 + 255) / 256, 256, 0, stream>>>(mask3, mask4, D3,H3,W3, D4,H4,W4);
  conv(16, bufB, bufA, mask4, nullptr, D3,H3,W3, D4,H4,W4, 2);
  // conv4 block (emit)
  conv(10, bufA, bufB, mask4, nullptr, D4,H4,W4, D4,H4,W4, 1);
  conv(11, bufB, bufA, mask4, nullptr, D4,H4,W4, D4,H4,W4, 1);
  conv(12, bufA, bufB, mask4, emit3,  D4,H4,W4, D4,H4,W4, 1);
}
